// WaveTFTForecaster_37065567764971
// MI455X (gfx1250) — compile-verified, matrix operand order checked by reading, not run on hardware
//
#include <hip/hip_runtime.h>
#include <hip/hip_bf16.h>

typedef __attribute__((ext_vector_type(16))) _Float16 v16h;
typedef __attribute__((ext_vector_type(4)))  _Float16 v4h;
typedef __attribute__((ext_vector_type(8)))  float    v8f;

#define Bm 256
#define Lm 128
#define Dm 256
#define Hm 8
#define DKm 32
#define NLm 6
#define Em 8
#define FFm 1024
#define NHm 5
#define NMOEm 3
#define NTOK (Bm * Lm)   // 32768

#define GEMM_GELU 1
#define GEMM_ACC  2
#define GEMM_ST16 4
#define GEMM_RS   8

// Tensor Data Mover path: probe-confirmed 5-arg builtin on ROCm 7.2 (clang-22).
#if __has_builtin(__builtin_amdgcn_tensor_load_to_lds) && \
    __has_builtin(__builtin_amdgcn_s_wait_tensorcnt) && (__clang_major__ < 23)
#define USE_TDM 1
#else
#define USE_TDM 0
#endif

#if USE_TDM
typedef unsigned int u32x4 __attribute__((ext_vector_type(4)));
typedef int          i32x8 __attribute__((ext_vector_type(8)));
typedef int          i32x4 __attribute__((ext_vector_type(4)));

// 2D f16 tile load via TDM: rows x cols elements, row stride in elements.
// D# packing per CDNA5 ISA (group0: count/lds/global/type,
// group1: data_size=2B, tensor dims, tile dims, dim0 stride).
__device__ __forceinline__ void tdm_load_2d_f16(
    const _Float16* gsrc, unsigned lds_off, unsigned rows, unsigned cols,
    unsigned long long row_stride_elems)
{
  unsigned long long ga = (unsigned long long)(uintptr_t)gsrc;
  u32x4 g0;
  g0[0] = 1u;                                   // count=1, user descriptor
  g0[1] = lds_off;                              // lds_addr (bytes)
  g0[2] = (unsigned)ga;                         // global_addr[31:0]
  g0[3] = (unsigned)(ga >> 32) | (2u << 30);    // global_addr[56:32] | type=2
  unsigned td0 = cols, td1 = rows;
  unsigned long long s0 = row_stride_elems;
  i32x8 g1;
  g1[0] = (int)(1u << 16);                      // data_size = 1 (2 bytes)
  g1[1] = (int)((td0 & 0xFFFFu) << 16);         // tensor_dim0[15:0]
  g1[2] = (int)((td0 >> 16) | ((td1 & 0xFFFFu) << 16));
  g1[3] = (int)((td1 >> 16) | (cols << 16));    // tile_dim0 = cols
  g1[4] = (int)rows;                            // tile_dim1 = rows, tile_dim2=0
  g1[5] = (int)(unsigned)(s0 & 0xFFFFFFFFull);  // dim0 stride lo
  g1[6] = (int)(unsigned)(s0 >> 32);            // dim0 stride hi
  g1[7] = 0;
  i32x4 z4 = {0, 0, 0, 0};
  __builtin_amdgcn_tensor_load_to_lds(g0, g1, z4, z4, 0);
}
#endif

__device__ __forceinline__ float gelu_exact(float x) {
  return 0.5f * x * (1.0f + erff(x * 0.70710678118654752f));
}

// CDNA5 16-bit 16x32 fragment map inverse: k -> half=(k>>3)&1 ; i=(k&7)+((k>>4)<<3)

// ---------------------------------------------------------------------------
// Generic WMMA GEMM, templated (branch-free staging + epilogue).
// 128x64 tile / 256 threads (8 waves); wave w: rows [16w,16w+16) x 64 cols.
// ---------------------------------------------------------------------------
template <bool A16P, bool HASB, int FLAGS>
__global__ __launch_bounds__(256) void gemm_wmma_kernel(
    const float* __restrict__ A32, const _Float16* __restrict__ A16,
    const float* __restrict__ W, const float* __restrict__ bias,
    float* __restrict__ Cf, _Float16* __restrict__ Ch,
    const float* __restrict__ rowscale, int rs_idx, int K, int M)
{
  __shared__ __align__(16) _Float16 Afrag[8][32][24];  // [rowblock][lane][i]
  __shared__ __align__(16) _Float16 Bfrag[4][32][24];  // [ctile][lane][i]

  const int tid  = threadIdx.x;
  const int wave = tid >> 5;
  const int lane = tid & 31;
  const int half = lane >> 4;
  const int ln   = lane & 15;
  const int row0 = blockIdx.y * 128;
  const int m0   = blockIdx.x * 64;

  v8f acc[4] = {};

  for (int k0 = 0; k0 < K; k0 += 32) {
    // ---- stage A tile (128x32) in fragment order; one b64 LDS store / 4 elems
#pragma unroll
    for (int u = 0; u < 4; ++u) {
      int it  = tid + u * 256;           // 0..1023
      int r   = it >> 3;                 // row 0..127
      int col = (it & 7) << 2;           // k offset 0,4,...,28
      int h   = (col >> 3) & 1;
      int i0  = (col & 7) + ((col >> 4) << 3);
      int tl  = (r & 15) + (h << 4);
      v4h hv;
      if constexpr (A16P) {
        hv = *(const v4h*)&A16[(size_t)(row0 + r) * K + (k0 + col)];
      } else {
        const float4 v = *(const float4*)&A32[(size_t)(row0 + r) * K + (k0 + col)];
        hv[0] = (_Float16)v.x; hv[1] = (_Float16)v.y;
        hv[2] = (_Float16)v.z; hv[3] = (_Float16)v.w;
      }
      *(v4h*)&Afrag[r >> 4][tl][i0] = hv;
    }
    // ---- stage B tile (32x64) in fragment order (scatter across 4 lanes)
#pragma unroll
    for (int u = 0; u < 2; ++u) {
      int it  = tid + u * 256;           // 0..511
      int k   = it >> 4;                 // 0..31
      int col = (it & 15) << 2;          // m offset 0,4,...,60
      const float4 v = *(const float4*)&W[(size_t)(k0 + k) * M + (m0 + col)];
      int h  = (k >> 3) & 1;
      int i  = (k & 7) + ((k >> 4) << 3);
      int ct = col >> 4;
      int n0 = (col & 15) + (h << 4);
      Bfrag[ct][n0 + 0][i] = (_Float16)v.x;
      Bfrag[ct][n0 + 1][i] = (_Float16)v.y;
      Bfrag[ct][n0 + 2][i] = (_Float16)v.z;
      Bfrag[ct][n0 + 3][i] = (_Float16)v.w;
    }
    if (k0 + 32 < K) {
      if constexpr (A16P) __builtin_prefetch(&A16[(size_t)(row0 + (tid >> 1)) * K + k0 + 32], 0, 1);
      else                __builtin_prefetch(&A32[(size_t)(row0 + (tid >> 1)) * K + k0 + 32], 0, 1);
    }
    __syncthreads();

    v16h a = *(const v16h*)Afrag[wave][lane];
#pragma unroll
    for (int c = 0; c < 4; ++c) {
      v16h b = *(const v16h*)Bfrag[c][lane];
      acc[c] = __builtin_amdgcn_wmma_f32_16x16x32_f16(
          false, a, false, b, (short)0, acc[c], false, false);
    }
    __syncthreads();
  }

  // epilogue: C layout — VGPR r holds rows (r, r+8) for half (0,1), col = ln
#pragma unroll
  for (int c = 0; c < 4; ++c) {
#pragma unroll
    for (int r = 0; r < 8; ++r) {
      int grow = row0 + wave * 16 + r + half * 8;
      int gcol = m0 + c * 16 + ln;
      float v = acc[c][r];
      if constexpr (HASB) v += bias[gcol];
      if constexpr (FLAGS & GEMM_GELU) v = gelu_exact(v);
      if constexpr (FLAGS & GEMM_RS)   v *= rowscale[(size_t)grow * Em + rs_idx];
      size_t oi = (size_t)grow * M + gcol;
      if constexpr (FLAGS & GEMM_ST16)     Ch[oi] = (_Float16)v;
      else if constexpr (FLAGS & GEMM_ACC) Cf[oi] += v;
      else                                 Cf[oi] = v;
    }
  }
}

// ---------------------------------------------------------------------------
// Fused wave-modulated attention; one workgroup per (batch, head).
// Q/K/V tiles arrive via TDM (tensor_load_to_lds) when available; all WMMA
// operands kept in LDS in fragment order.
// ---------------------------------------------------------------------------
__global__ __launch_bounds__(256) void attn_kernel(
    const _Float16* __restrict__ Q, const _Float16* __restrict__ Kh,
    const _Float16* __restrict__ V, _Float16* __restrict__ O,
    const float* __restrict__ wfreq, const float* __restrict__ wphase, int layer)
{
  // PF (49152 B) overlaps QF (12288) + KF (12288) + raw TDM window (24576);
  // VF above: total 61440 B static LDS.
  __shared__ __align__(16) unsigned char smem[61440];
  _Float16 (*PF)[4][32][24] = (_Float16 (*)[4][32][24])smem;          // [8][4][32][24]
  _Float16 (*QF)[32][24]    = (_Float16 (*)[32][24])smem;             // [8][32][24]
  _Float16 (*KF)[32][24]    = (_Float16 (*)[32][24])(smem + 12288);   // [8][32][24]
  _Float16 (*VF)[2][32][24] = (_Float16 (*)[2][32][24])(smem + 49152);// [4][2][32][24]

  const int head  = blockIdx.x & 7;
  const int batch = blockIdx.x >> 3;
  const int tid  = threadIdx.x;
  const int wave = tid >> 5;
  const int lane = tid & 31;
  const int half = lane >> 4;
  const int ln   = lane & 15;

  const size_t base = ((size_t)batch * Lm) * Dm + head * DKm;

#if USE_TDM
  _Float16* rawQ = (_Float16*)(smem + 24576);   // [128][32] raw tiles
  _Float16* rawK = rawQ + 4096;
  _Float16* rawV = rawK + 4096;
  if (wave == 0) {
    tdm_load_2d_f16(Q  + base, (unsigned)(uintptr_t)rawQ, Lm, DKm, Dm);
    tdm_load_2d_f16(Kh + base, (unsigned)(uintptr_t)rawK, Lm, DKm, Dm);
    tdm_load_2d_f16(V  + base, (unsigned)(uintptr_t)rawV, Lm, DKm, Dm);
    __builtin_amdgcn_s_wait_tensorcnt(0);
  }
  __syncthreads();
#endif

  // ---- stage Q,K (A-frag layout) and V (B-frag layout)
#pragma unroll
  for (int u = 0; u < 4; ++u) {
    int it  = tid + u * 256;         // 0..1023
    int r   = it >> 3;               // key/query row 0..127
    int col = (it & 7) << 2;         // feature 0,4,...,28
#if USE_TDM
    v4h qv = *(const v4h*)&rawQ[r * DKm + col];
    v4h kv = *(const v4h*)&rawK[r * DKm + col];
    v4h vv = *(const v4h*)&rawV[r * DKm + col];
#else
    size_t g = base + (size_t)r * Dm + col;
    v4h qv = *(const v4h*)&Q[g];
    v4h kv = *(const v4h*)&Kh[g];
    v4h vv = *(const v4h*)&V[g];
#endif
    int h  = (col >> 3) & 1;
    int i0 = (col & 7) + ((col >> 4) << 3);
    int tl = (r & 15) + (h << 4);
    *(v4h*)&QF[r >> 4][tl][i0] = qv;
    *(v4h*)&KF[r >> 4][tl][i0] = kv;
    // V as B-fragments: k = r (key), n = col..col+3 (feature)
    int kc = r >> 5;
    int kl = r & 31;
    int vh = (kl >> 3) & 1;
    int vi = (kl & 7) + ((kl >> 4) << 3);
    int ct = col >> 4;
    int n0 = (col & 15) + (vh << 4);
    VF[kc][ct][n0 + 0][vi] = vv[0];
    VF[kc][ct][n0 + 1][vi] = vv[1];
    VF[kc][ct][n0 + 2][vi] = vv[2];
    VF[kc][ct][n0 + 3][vi] = vv[3];
  }
  __syncthreads();

  const float freq  = wfreq[layer * Hm + head];
  const float phase = wphase[layer * Hm + head];

  // ---- S = Q K^T : wave w owns query rows [16w,16w+16), all 128 key cols
  v16h a = *(const v16h*)QF[wave][lane];
  v8f s[8];
#pragma unroll
  for (int t = 0; t < 8; ++t) {
    v16h b = *(const v16h*)KF[t][lane];
    v8f z = {};
    s[t] = __builtin_amdgcn_wmma_f32_16x16x32_f16(
        false, a, false, b, (short)0, z, false, false);
  }

  const float scale = 0.17677669529663688f;  // 1/sqrt(32)
#pragma unroll
  for (int t = 0; t < 8; ++t) {
    int col = t * 16 + ln;
    // hardware v_cos_f32; args bounded (|x| <= 2*pi*0.1*128-ish)
    float f = scale * __cosf(6.283185307179586f * freq * (float)col + phase);
#pragma unroll
    for (int r = 0; r < 8; ++r) s[t][r] *= f;
  }

  __syncthreads();   // all waves done reading QF/KF before PF overwrites them

  // ---- softmax per row; lanes [0,15] / [16,31] of one VGPR hold one row
#pragma unroll
  for (int r = 0; r < 8; ++r) {
    float mx = -1e30f;
#pragma unroll
    for (int t = 0; t < 8; ++t) mx = fmaxf(mx, s[t][r]);
    mx = fmaxf(mx, __shfl_xor(mx, 1, 32));
    mx = fmaxf(mx, __shfl_xor(mx, 2, 32));
    mx = fmaxf(mx, __shfl_xor(mx, 4, 32));
    mx = fmaxf(mx, __shfl_xor(mx, 8, 32));
    float sum = 0.0f;
#pragma unroll
    for (int t = 0; t < 8; ++t) { float e = __expf(s[t][r] - mx); s[t][r] = e; sum += e; }
    sum += __shfl_xor(sum, 1, 32);
    sum += __shfl_xor(sum, 2, 32);
    sum += __shfl_xor(sum, 4, 32);
    sum += __shfl_xor(sum, 8, 32);
    float inv = 1.0f / sum;
    // write P directly in A-fragment order: k = t*16+ln, m = r + 8*half
    int tl = r + 8 * half + ((ln >> 3) << 4);
#pragma unroll
    for (int t = 0; t < 8; ++t) {
      PF[wave][t >> 1][tl][(ln & 7) + ((t & 1) << 3)] = (_Float16)(s[t][r] * inv);
    }
  }
  __syncthreads();

  // ---- O = P V : rows [16w,16w+16) x 32 cols, K = 128 in 4 chunks
  v8f o0 = {}, o1 = {};
#pragma unroll
  for (int kc = 0; kc < 4; ++kc) {
    v16h pa = *(const v16h*)PF[wave][kc][lane];
    v16h b0 = *(const v16h*)VF[kc][0][lane];
    v16h b1 = *(const v16h*)VF[kc][1][lane];
    o0 = __builtin_amdgcn_wmma_f32_16x16x32_f16(false, pa, false, b0, (short)0, o0, false, false);
    o1 = __builtin_amdgcn_wmma_f32_16x16x32_f16(false, pa, false, b1, (short)0, o1, false, false);
  }

#pragma unroll
  for (int r = 0; r < 8; ++r) {
    int row = wave * 16 + r + half * 8;
    size_t g = base + (size_t)row * Dm;
    O[g + ln]      = (_Float16)o0[r];
    O[g + 16 + ln] = (_Float16)o1[r];
  }
}

// ---------------------------------------------------------------------------
// Small VALU kernels
// ---------------------------------------------------------------------------
__global__ void embed_kernel(const float* __restrict__ x,
                             const float* __restrict__ ew,
                             const float* __restrict__ eb,
                             float* __restrict__ h) {
  int i = blockIdx.x * 256 + threadIdx.x;
  if (i >= NTOK * Dm) return;
  int tok = i >> 8, d = i & 255;
  const float* xp = x + (size_t)tok * 6;
  float acc = eb[d];
#pragma unroll
  for (int j = 0; j < 6; ++j) acc += xp[j] * ew[j * Dm + d];
  h[i] = acc;
}

__global__ void zero_kernel(float* __restrict__ p, int n) {
  int i = blockIdx.x * 256 + threadIdx.x;
  if (i < n) p[i] = 0.0f;
}

__global__ void add_kernel(float* __restrict__ a, const float* __restrict__ b, int n) {
  int i = blockIdx.x * 256 + threadIdx.x;
  if (i < n) a[i] += b[i];
}

__global__ void router_logits_kernel(const float* __restrict__ h,
                                     const float* __restrict__ rw,
                                     const float* __restrict__ rb,
                                     float* __restrict__ logits) {
  int i = blockIdx.x * 256 + threadIdx.x;
  if (i >= NTOK * Em) return;
  int tok = i >> 3, e = i & 7;
  const float* hp = h + (size_t)tok * Dm;
  float acc = rb[e];
  for (int d = 0; d < Dm; ++d) acc += hp[d] * rw[d * Em + e];
  logits[i] = acc;
}

__global__ void combine_kernel(const float* __restrict__ logits,
                               float* __restrict__ combine) {
  int tok = blockIdx.x * 256 + threadIdx.x;
  if (tok >= NTOK) return;
  float l[Em], mx = -1e30f;
#pragma unroll
  for (int e = 0; e < Em; ++e) { l[e] = logits[(size_t)tok * Em + e]; mx = fmaxf(mx, l[e]); }
  float sum = 0.0f;
#pragma unroll
  for (int e = 0; e < Em; ++e) { l[e] = __expf(l[e] - mx); sum += l[e]; }
  float inv = 1.0f / sum;
#pragma unroll
  for (int e = 0; e < Em; ++e) l[e] *= inv;
  int i1 = 0;
#pragma unroll
  for (int e = 1; e < Em; ++e) if (l[e] > l[i1]) i1 = e;
  int i2 = (i1 == 0) ? 1 : 0;
#pragma unroll
  for (int e = 0; e < Em; ++e) if (e != i1 && l[e] > l[i2]) i2 = e;
  float tot = l[i1] + l[i2];
  float w1 = l[i1] / tot, w2 = l[i2] / tot;
#pragma unroll
  for (int e = 0; e < Em; ++e) {
    combine[(size_t)tok * Em + e] = (e == i1) ? w1 : ((e == i2) ? w2 : 0.0f);
  }
}

// last-token LayerNorm + prediction heads; one block per batch element
__global__ __launch_bounds__(256) void head_kernel(
    const float* __restrict__ h, const float* __restrict__ g,
    const float* __restrict__ beta, const float* __restrict__ pw,
    const float* __restrict__ pb, const float* __restrict__ uw,
    const float* __restrict__ ub, float* __restrict__ out) {
  __shared__ float rbuf[256];
  __shared__ float hn[256];
  int b = blockIdx.x;
  int d = threadIdx.x;
  size_t t = ((size_t)b * Lm + (Lm - 1)) * Dm;
  float v = h[t + d];

  rbuf[d] = v; __syncthreads();
  for (int s = 128; s > 0; s >>= 1) { if (d < s) rbuf[d] += rbuf[d + s]; __syncthreads(); }
  float mu = rbuf[0] * (1.0f / Dm); __syncthreads();
  float dv = v - mu;
  rbuf[d] = dv * dv; __syncthreads();
  for (int s = 128; s > 0; s >>= 1) { if (d < s) rbuf[d] += rbuf[d + s]; __syncthreads(); }
  float var = rbuf[0] * (1.0f / Dm);
  float r = rsqrtf(var + 1e-5f);
  hn[d] = dv * r * g[d] + beta[d];
  __syncthreads();

  if (d < NHm) {
    float acc = pb[d];
    for (int k = 0; k < Dm; ++k) acc += hn[k] * pw[k * NHm + d];
    out[(size_t)b * NHm + d] = acc;
  } else if (d >= 8 && d < 8 + NHm) {
    int j = d - 8;
    float acc = ub[j];
    for (int k = 0; k < Dm; ++k) acc += hn[k] * uw[k * NHm + j];
    out[(size_t)Bm * NHm + (size_t)b * NHm + j] =
        (acc > 20.0f) ? acc : log1pf(__expf(acc));
  }
}

// ---------------------------------------------------------------------------
extern "C" void kernel_launch(void* const* d_in, const int* in_sizes, int n_in,
                              void* d_out, int out_size, void* d_ws, size_t ws_size,
                              hipStream_t stream) {
  const float* x        = (const float*)d_in[0];
  const float* emb_w    = (const float*)d_in[1];
  const float* emb_b    = (const float*)d_in[2];
  const float* qw       = (const float*)d_in[3];
  const float* kw       = (const float*)d_in[4];
  const float* vw       = (const float*)d_in[5];
  const float* ow       = (const float*)d_in[6];
  const float* wfreq    = (const float*)d_in[7];
  const float* wphase   = (const float*)d_in[8];
  const float* router_w = (const float*)d_in[9];
  const float* router_b = (const float*)d_in[10];
  const float* e_w1     = (const float*)d_in[11];
  const float* e_b1     = (const float*)d_in[12];
  const float* e_w2     = (const float*)d_in[13];
  const float* e_b2     = (const float*)d_in[14];
  const float* ln_g     = (const float*)d_in[15];
  const float* ln_b     = (const float*)d_in[16];
  const float* pred_w   = (const float*)d_in[17];
  const float* pred_b   = (const float*)d_in[18];
  const float* unc_w    = (const float*)d_in[19];
  const float* unc_b    = (const float*)d_in[20];
  float* out = (float*)d_out;

  char* ws = (char*)d_ws;
  float*     h       = (float*)ws;      ws += (size_t)NTOK * Dm * 4;   // 33.5 MB
  _Float16*  q16     = (_Float16*)ws;   ws += (size_t)NTOK * Dm * 2;
  _Float16*  k16     = (_Float16*)ws;   ws += (size_t)NTOK * Dm * 2;
  _Float16*  v16     = (_Float16*)ws;   ws += (size_t)NTOK * Dm * 2;
  _Float16*  attn16  = (_Float16*)ws;   ws += (size_t)NTOK * Dm * 2;
  float*     logits  = (float*)ws;      ws += (size_t)NTOK * Em * 4;
  float*     combine = (float*)ws;      ws += (size_t)NTOK * Em * 4;
  _Float16*  ff16    = (_Float16*)ws;   ws += (size_t)NTOK * FFm * 2;  // 67 MB
  float*     moe_out = (float*)ws;      ws += (size_t)NTOK * Dm * 4;

  const int elemBlocks = (NTOK * Dm + 255) / 256;
  embed_kernel<<<elemBlocks, 256, 0, stream>>>(x, emb_w, emb_b, h);

  const dim3 gProj(Dm / 64, NTOK / 128);   // (4, 256)
  const dim3 gFF1(FFm / 64, NTOK / 128);   // (16, 256)

  for (int i = 0; i < NLm; ++i) {
    const float* qwi = qw + (size_t)i * Dm * Dm;
    const float* kwi = kw + (size_t)i * Dm * Dm;
    const float* vwi = vw + (size_t)i * Dm * Dm;
    const float* owi = ow + (size_t)i * Dm * Dm;

    gemm_wmma_kernel<false, false, GEMM_ST16><<<gProj, 256, 0, stream>>>(
        h, nullptr, qwi, nullptr, nullptr, q16, nullptr, 0, Dm, Dm);
    gemm_wmma_kernel<false, false, GEMM_ST16><<<gProj, 256, 0, stream>>>(
        h, nullptr, kwi, nullptr, nullptr, k16, nullptr, 0, Dm, Dm);
    gemm_wmma_kernel<false, false, GEMM_ST16><<<gProj, 256, 0, stream>>>(
        h, nullptr, vwi, nullptr, nullptr, v16, nullptr, 0, Dm, Dm);

    attn_kernel<<<Bm * Hm, 256, 0, stream>>>(q16, k16, v16, attn16, wfreq, wphase, i);

    // h += attn @ ow  (residual fused as accumulate)
    gemm_wmma_kernel<true, false, GEMM_ACC><<<gProj, 256, 0, stream>>>(
        nullptr, attn16, owi, nullptr, h, nullptr, nullptr, 0, Dm, Dm);

    if ((i % 2 == 0) && (i / 2 < NMOEm)) {
      int j = i / 2;
      router_logits_kernel<<<(NTOK * Em + 255) / 256, 256, 0, stream>>>(
          h, router_w + (size_t)j * Dm * Em, router_b + (size_t)j * Em, logits);
      combine_kernel<<<(NTOK + 255) / 256, 256, 0, stream>>>(logits, combine);
      zero_kernel<<<elemBlocks, 256, 0, stream>>>(moe_out, NTOK * Dm);

      for (int e = 0; e < Em; ++e) {
        const float* w1 = e_w1 + ((size_t)j * Em + e) * Dm * FFm;
        const float* b1 = e_b1 + ((size_t)j * Em + e) * FFm;
        const float* w2 = e_w2 + ((size_t)j * Em + e) * FFm * Dm;
        const float* b2 = e_b2 + ((size_t)j * Em + e) * Dm;
        gemm_wmma_kernel<false, true, GEMM_ST16 | GEMM_GELU><<<gFF1, 256, 0, stream>>>(
            h, nullptr, w1, b1, nullptr, ff16, nullptr, 0, Dm, FFm);
        gemm_wmma_kernel<true, true, GEMM_ACC | GEMM_RS><<<gProj, 256, 0, stream>>>(
            nullptr, ff16, w2, b2, moe_out, nullptr, combine, e, FFm, Dm);
      }
      add_kernel<<<elemBlocks, 256, 0, stream>>>(h, moe_out, NTOK * Dm);
    }
  }

  head_kernel<<<Bm, 256, 0, stream>>>(h, ln_g, ln_b, pred_w, pred_b, unc_w, unc_b, out);
}
